// MultiHeadAttention_37477884625063
// MI455X (gfx1250) — matrix operand order repaired and checked by reading, not compiled, not run on hardware
//
#include <hip/hip_runtime.h>
#include <hip/hip_bf16.h>

#define BATCH 256
#define SEQ   256
#define EMB   384
#define NHEAD 6
#define HDIM  64
#define NQKV  1152   // 3*EMB, packed QKV columns

typedef __bf16 bf16;
typedef __attribute__((ext_vector_type(16))) __bf16 v16bf;
typedef __attribute__((ext_vector_type(8)))  float  v8f;

static __device__ __forceinline__ v8f wmma_bf16(v16bf a, v16bf b, v8f c) {
  // D = A(16x32 bf16) x B(32x16 bf16) + C(16x16 f32)
  return __builtin_amdgcn_wmma_f32_16x16x32_bf16(false, a, false, b, (short)0, c,
                                                 false, false);
}

// A-matrix fragment (16x32 bf16) from row-major [rows][ld] source.
// ISA layout: lane m=lane&15 holds row m; half=lane>>4 selects K blocks:
// elems 0..7 -> K = half*8 + e ; elems 8..15 -> K = 16 + half*8 + (e-8)
// Two 16B contiguous loads per lane (ld must be a multiple of 8, base 16B-aligned).
static __device__ __forceinline__ v16bf fragA_ld(const bf16* base, int ld) {
  const int lane = threadIdx.x & 31;
  const int m = lane & 15, half = lane >> 4;
  const bf16* p = base + m * ld + half * 8;
  union { int4 q[2]; v16bf v; } u;
  u.q[0] = *(const int4*)p;
  u.q[1] = *(const int4*)(p + 16);
  return u.v;
}

// B-matrix fragment (32x16 bf16) where the SOURCE stores B transposed,
// row-major [N][ld(K)] (so B[K][n] = src[n][K]).  ISA layout: lane n=lane&15
// holds column n; elems 0..7 -> K = half*8 + e ; elems 8..15 -> K = 16 + half*8 + (e-8).
static __device__ __forceinline__ v16bf fragB_ldT(const bf16* base, int ld) {
  const int lane = threadIdx.x & 31;
  const int n = lane & 15, half = lane >> 4;
  const bf16* p = base + n * ld + half * 8;
  union { int4 q[2]; v16bf v; } u;
  u.q[0] = *(const int4*)p;
  u.q[1] = *(const int4*)(p + 16);
  return u.v;
}

// ---------------------------------------------------------------- weights prep
__global__ void prep_weights(const float* __restrict__ Wq, const float* __restrict__ Wk,
                             const float* __restrict__ Wv, const float* __restrict__ Wo,
                             bf16* __restrict__ Wpack, bf16* __restrict__ Wo_bf) {
  const int idx = blockIdx.x * blockDim.x + threadIdx.x;
  const int stride = gridDim.x * blockDim.x;
  const int total1 = EMB * NQKV;
  for (int i = idx; i < total1; i += stride) {
    const int e = i / NQKV;
    const int c = i % NQKV;
    const int proj = c / EMB;
    const int rem = c % EMB;
    const int h = rem / HDIM;
    const int d = rem % HDIM;
    const float* W = (proj == 0) ? Wq : (proj == 1 ? Wk : Wv);
    Wpack[i] = (bf16)W[((size_t)h * EMB + e) * HDIM + d];  // [H,E,D] -> [E, 3*H*D]
  }
  const int total2 = EMB * EMB;
  for (int i = idx; i < total2; i += stride) Wo_bf[i] = (bf16)Wo[i];
}

// ----------------------------------------------------------------- QKV GEMM
// C[65536 x 1152] = x[65536 x 384] @ Wpack[384 x 1152], bf16 out scattered to
// qkv[proj][b][h][t][d].  Block tile 256(M) x 64(N); wave tile 32 x 64.
__global__ __launch_bounds__(256)
void qkv_gemm(const float* __restrict__ x, const bf16* __restrict__ Wpack,
              bf16* __restrict__ qkv) {
  __shared__ bf16 As[256 * 32];   // 16 KB, row-major [m][k]
  __shared__ bf16 Bs_t[64 * 32];  //  4 KB, transposed [n][k]
  const int mt = blockIdx.x;      // 0..255
  const int nt = blockIdx.y;      // 0..17 (one (proj, head) per 64-col tile)
  const int tid = threadIdx.x;
  const int wave = tid >> 5;
  v8f acc[2][4] = {};

  for (int k0 = 0; k0 < EMB; k0 += 32) {
    // stage A: 256x32 f32 -> bf16 (float4 loads, int2 LDS stores)
#pragma unroll
    for (int it = 0; it < 8; ++it) {
      const int i = tid + it * 256;
      const int r = i >> 3;
      const int c0 = (i & 7) * 4;
      const float4 xv = *(const float4*)&x[(size_t)(mt * 256 + r) * EMB + k0 + c0];
      union { bf16 e[4]; int2 v; } u;
      u.e[0] = (bf16)xv.x; u.e[1] = (bf16)xv.y;
      u.e[2] = (bf16)xv.z; u.e[3] = (bf16)xv.w;
      *(int2*)&As[r * 32 + c0] = u.v;
    }
    // stage B transposed: [32][64] global -> [64][32] LDS (int4 loads)
    {
      const int r = tid >> 3;        // k row 0..31
      const int c0 = (tid & 7) * 8;  // col group
      union { int4 v; bf16 e[8]; } u;
      u.v = *(const int4*)&Wpack[(size_t)(k0 + r) * NQKV + nt * 64 + c0];
#pragma unroll
      for (int e = 0; e < 8; ++e) Bs_t[(c0 + e) * 32 + r] = u.e[e];
    }
    __syncthreads();
    const v16bf a0 = fragA_ld(As + (wave * 32 + 0) * 32, 32);
    const v16bf a1 = fragA_ld(As + (wave * 32 + 16) * 32, 32);
#pragma unroll
    for (int j = 0; j < 4; ++j) {
      const v16bf b = fragB_ldT(Bs_t + j * 16 * 32, 32);
      acc[0][j] = wmma_bf16(a0, b, acc[0][j]);
      acc[1][j] = wmma_bf16(a1, b, acc[1][j]);
    }
    __syncthreads();
  }

  const int lane = tid & 31;
  const int n = lane & 15, mbase = (lane >> 4) * 8;
  const int proj = nt / NHEAD, h = nt % NHEAD;
  const size_t projbase = (size_t)proj * BATCH * NHEAD * SEQ * HDIM;
#pragma unroll
  for (int s = 0; s < 2; ++s)
#pragma unroll
    for (int j = 0; j < 4; ++j)
#pragma unroll
      for (int r = 0; r < 8; ++r) {
        const int row = mt * 256 + wave * 32 + s * 16 + mbase + r;
        const int b_ = row >> 8, t = row & 255;
        const int d = j * 16 + n;
        qkv[projbase + ((size_t)b_ * NHEAD + h) * (SEQ * HDIM) + (size_t)t * HDIM + d]
            = (bf16)acc[s][j][r];
      }
}

// ------------------------------------------------------------ flash attention
// One workgroup per (b,h).  V staged TRANSPOSED in LDS ([d][t]) so the PV
// B-fragments are contiguous; K read directly from global (k^T B-fragment is
// two contiguous 16B loads per lane).  Online softmax over 32-key pairs; P routed
// wave-locally through LDS with s_wait_dscnt (divergent causal trip counts).
__global__ __launch_bounds__(256)
void flash_attn(const bf16* __restrict__ qkv, bf16* __restrict__ attnout) {
  __shared__ bf16 Vs_t[HDIM * SEQ];  // 32 KB, [d][t]
  __shared__ bf16 Pbuf[8][16 * 32];  //  8 KB (per-wave P tiles, row-major [m][k])
  const int bh = blockIdx.x;
  const int b = bh / NHEAD, h = bh % NHEAD;
  const int tid = threadIdx.x;
  const int wave = tid >> 5;
  const int lane = tid & 31;
  const int n = lane & 15, mbase = (lane >> 4) * 8;

  const size_t plane = (size_t)SEQ * HDIM;
  const bf16* qh = qkv + (size_t)bh * plane;
  const bf16* kh = qkv + ((size_t)BATCH * NHEAD + bh) * plane;
  const bf16* vh = qkv + ((size_t)2 * BATCH * NHEAD + bh) * plane;

  // stage V transposed: int4 global loads, bf16 scatter into LDS
#pragma unroll
  for (int it = 0; it < 8; ++it) {
    const int i = tid + it * 256;      // 2048 int4 chunks
    const int t = i >> 3;
    const int d0 = (i & 7) * 8;
    union { int4 v; bf16 e[8]; } u;
    u.v = ((const int4*)vh)[i];
#pragma unroll
    for (int e = 0; e < 8; ++e) Vs_t[(d0 + e) * SEQ + t] = u.e[e];
  }
  __syncthreads();

  const float scale = 0.125f;  // 1/sqrt(64)
  bf16* Pw = &Pbuf[wave][0];

  for (int qiter = 0; qiter < 2; ++qiter) {
    const int qt = qiter * 8 + wave;  // query tile 0..15
    const int qbase = qt * 16;
    const v16bf qa0 = fragA_ld(qh + (size_t)qbase * HDIM, HDIM);       // d 0..31
    const v16bf qa1 = fragA_ld(qh + (size_t)qbase * HDIM + 32, HDIM);  // d 32..63

    v8f acc[4] = {};
    float rmax[8], rsum[8];
#pragma unroll
    for (int r = 0; r < 8; ++r) { rmax[r] = -3.0e38f; rsum[r] = 0.0f; }

    const int jmax = (qbase + 15) >> 5;  // causal: last 32-key pair touched
    for (int jj = 0; jj <= jmax; ++jj) {
      const int key0 = jj * 32;
      // scores tile 0: keys key0..key0+15   (B = k^T, contiguous loads)
      const v16bf kb0a = fragB_ldT(kh + (size_t)key0 * HDIM, HDIM);
      const v16bf kb0b = fragB_ldT(kh + (size_t)key0 * HDIM + 32, HDIM);
      v8f s0 = {};
      s0 = wmma_bf16(qa0, kb0a, s0);
      s0 = wmma_bf16(qa1, kb0b, s0);
      // scores tile 1: keys key0+16..key0+31 (may be fully masked)
      const bool tile1 = (key0 + 16) <= (qbase + 15);
      v8f s1 = {};
      if (tile1) {
        const v16bf kb1a = fragB_ldT(kh + (size_t)(key0 + 16) * HDIM, HDIM);
        const v16bf kb1b = fragB_ldT(kh + (size_t)(key0 + 16) * HDIM + 32, HDIM);
        s1 = wmma_bf16(qa0, kb1a, s1);
        s1 = wmma_bf16(qa1, kb1b, s1);
      }
      // masked online-softmax update (C layout: row = qbase+mbase+r, col n)
#pragma unroll
      for (int r = 0; r < 8; ++r) {
        const int qrow = qbase + mbase + r;
        float v0 = s0[r] * scale;
        if (key0 + n > qrow) v0 = -3.0e38f;
        float v1 = -3.0e38f;
        if (tile1) {
          v1 = s1[r] * scale;
          if (key0 + 16 + n > qrow) v1 = -3.0e38f;
        }
        float m = fmaxf(v0, v1);
        m = fmaxf(m, __shfl_xor(m, 1));
        m = fmaxf(m, __shfl_xor(m, 2));
        m = fmaxf(m, __shfl_xor(m, 4));
        m = fmaxf(m, __shfl_xor(m, 8));
        const float nm = fmaxf(rmax[r], m);
        const float alpha = __expf(rmax[r] - nm);
        rmax[r] = nm;
        const float p0 = __expf(v0 - nm);
        const float p1 = tile1 ? __expf(v1 - nm) : 0.0f;
        float ps = p0 + p1;
        ps += __shfl_xor(ps, 1);
        ps += __shfl_xor(ps, 2);
        ps += __shfl_xor(ps, 4);
        ps += __shfl_xor(ps, 8);
        rsum[r] = rsum[r] * alpha + ps;
#pragma unroll
        for (int j = 0; j < 4; ++j) acc[j][r] *= alpha;
        Pw[(mbase + r) * 32 + n] = (bf16)p0;
        Pw[(mbase + r) * 32 + 16 + n] = (bf16)p1;
      }
      // wave-local LDS RAW fence (waves diverge -> no block barrier; per-wave
      // DS ops are in-order, fence the compiler + counter)
      __asm__ volatile("s_wait_dscnt 0" ::: "memory");
      const v16bf pa = fragA_ld(Pw, 32);  // 16x32 P tile as A operand
#pragma unroll
      for (int j = 0; j < 4; ++j) {
        const v16bf vb = fragB_ldT(Vs_t + (size_t)(j * 16) * SEQ + key0, SEQ);
        acc[j] = wmma_bf16(pa, vb, acc[j]);
      }
      __asm__ volatile("s_wait_dscnt 0" ::: "memory");
    }
    // normalize rows and store concat-head layout [b][t][h*64+d] in bf16
#pragma unroll
    for (int j = 0; j < 4; ++j) {
#pragma unroll
      for (int r = 0; r < 8; ++r) {
        const int t = qbase + mbase + r;
        const int d = j * 16 + n;
        const float o = acc[j][r] / rsum[r];
        attnout[((size_t)b * SEQ + t) * EMB + h * HDIM + d] = (bf16)o;
      }
    }
  }
}

// ------------------------------------------------------------- output proj
// out[65536 x 384] = attn[65536 x 384] @ Wo[384 x 384] + bo, f32 output.
__global__ __launch_bounds__(256)
void out_proj(const bf16* __restrict__ A, const bf16* __restrict__ Wo_bf,
              const float* __restrict__ bo, float* __restrict__ out) {
  __shared__ bf16 As[256 * 32];   // 16 KB
  __shared__ bf16 Bs_t[64 * 32];  //  4 KB, transposed [n][k]
  const int mt = blockIdx.x;      // 0..255
  const int nt = blockIdx.y;      // 0..5
  const int tid = threadIdx.x;
  const int wave = tid >> 5;
  v8f acc[2][4] = {};

  for (int k0 = 0; k0 < EMB; k0 += 32) {
    // stage A: 256x32 bf16 copy (int4)
#pragma unroll
    for (int it = 0; it < 4; ++it) {
      const int i = tid + it * 256;    // 1024 int4 chunks
      const int r = i >> 2;
      const int c0 = (i & 3) * 8;
      *(int4*)&As[r * 32 + c0] =
          *(const int4*)&A[(size_t)(mt * 256 + r) * EMB + k0 + c0];
    }
    // stage B transposed
    {
      const int r = tid >> 3;
      const int c0 = (tid & 7) * 8;
      union { int4 v; bf16 e[8]; } u;
      u.v = *(const int4*)&Wo_bf[(size_t)(k0 + r) * EMB + nt * 64 + c0];
#pragma unroll
      for (int e = 0; e < 8; ++e) Bs_t[(c0 + e) * 32 + r] = u.e[e];
    }
    __syncthreads();
    const v16bf a0 = fragA_ld(As + (wave * 32 + 0) * 32, 32);
    const v16bf a1 = fragA_ld(As + (wave * 32 + 16) * 32, 32);
#pragma unroll
    for (int j = 0; j < 4; ++j) {
      const v16bf bfrag = fragB_ldT(Bs_t + j * 16 * 32, 32);
      acc[0][j] = wmma_bf16(a0, bfrag, acc[0][j]);
      acc[1][j] = wmma_bf16(a1, bfrag, acc[1][j]);
    }
    __syncthreads();
  }

  const int lane = tid & 31;
  const int n = lane & 15, mbase = (lane >> 4) * 8;
#pragma unroll
  for (int j = 0; j < 4; ++j) {
    const int col = nt * 64 + j * 16 + n;
    const float bias = bo[col];
#pragma unroll
    for (int s = 0; s < 2; ++s)
#pragma unroll
      for (int r = 0; r < 8; ++r) {
        const int row = mt * 256 + wave * 32 + s * 16 + mbase + r;
        out[(size_t)row * EMB + col] = acc[s][j][r] + bias;
      }
  }
}

extern "C" void kernel_launch(void* const* d_in, const int* in_sizes, int n_in,
                              void* d_out, int out_size, void* d_ws, size_t ws_size,
                              hipStream_t stream) {
  (void)in_sizes; (void)n_in; (void)out_size; (void)ws_size;
  const float* x  = (const float*)d_in[0];
  const float* Wq = (const float*)d_in[1];
  const float* Wk = (const float*)d_in[2];
  const float* Wv = (const float*)d_in[3];
  const float* Wo = (const float*)d_in[4];
  const float* bo = (const float*)d_in[5];
  float* out = (float*)d_out;

  // workspace layout (bytes, 256-aligned regions)
  char* ws = (char*)d_ws;
  bf16* Wpack = (bf16*)(ws);                  //   884,736 B  [E x 1152]
  bf16* Wo_bf = (bf16*)(ws + 884736);         //   294,912 B  [E x E]
  bf16* qkv   = (bf16*)(ws + 1179648);        // 150,994,944 B [3][B][H][T][D]
  bf16* attn  = (bf16*)(ws + 152174592);      //  50,331,648 B [B][T][E]

  prep_weights<<<256, 256, 0, stream>>>(Wq, Wk, Wv, Wo, Wpack, Wo_bf);
  qkv_gemm<<<dim3(256, 18), 256, 0, stream>>>(x, Wpack, qkv);
  flash_attn<<<dim3(BATCH * NHEAD), 256, 0, stream>>>(qkv, attn);
  out_proj<<<dim3(256, 6), 256, 0, stream>>>(attn, Wo_bf, bo, out);
}
